// MoVaELayer_84241488544006
// MI455X (gfx1250) — matrix operand order repaired and checked by reading, not compiled, not run on hardware
//
#include <hip/hip_runtime.h>
#include <stdint.h>

typedef __attribute__((ext_vector_type(16))) __bf16 v16bf;
typedef __attribute__((ext_vector_type(8)))  float  v8f;

namespace {
constexpr int kN    = 4096;   // B*T tokens
constexpr int kC    = 1024;   // model dim
constexpr int kH    = 2048;   // MLP hidden
constexpr int kEmlp = 8;
constexpr int kEtot = 12;
constexpr int kV    = 16384;
constexpr int kMt   = 32;     // token tile (rows per block) = 2 WMMA row-tiles
constexpr int kXS   = kC + 8; // padded LDS stride (halfs): bank-conflict-free frags
constexpr int kHS   = kH + 8; // padded LDS stride (halfs) for hidden tile
}

__device__ __forceinline__ unsigned short f32_to_bf16_rne(float f) {
  union { float f; unsigned u; } cv; cv.f = f;
  unsigned u = cv.u;
  unsigned r = (u + 0x7FFFu + ((u >> 16) & 1u)) >> 16;  // round-to-nearest-even
  return (unsigned short)r;
}

union FragBF { v16bf v; uint4 q[2]; };

// ---------------------------------------------------------------------------
// 1) Router: f32 scores, softmax, top-2, per-expert compaction lists.
//    One wave32 per token; router_w cached in LDS (48 KB). Exact f32 so the
//    top-k selection matches the reference bitwise.
// ---------------------------------------------------------------------------
__global__ __launch_bounds__(256) void router_kernel(
    const float* __restrict__ x, const float* __restrict__ router_w,
    float* __restrict__ rw_out, int* __restrict__ counts,
    int* __restrict__ list_packed, float* __restrict__ list_w,
    int* __restrict__ meta_idx, float* __restrict__ meta_w)
{
  __shared__ float s_rw[kEtot * kC];
  const float4* src = (const float4*)router_w;
  float4* dst = (float4*)s_rw;
  for (int i = threadIdx.x; i < kEtot * kC / 4; i += 256) dst[i] = src[i];
  __syncthreads();

  const int wave = threadIdx.x >> 5;
  const int lane = threadIdx.x & 31;
  const int n = blockIdx.x * 8 + wave;

  float p[kEtot];
  #pragma unroll
  for (int e = 0; e < kEtot; ++e) p[e] = 0.f;
  const float* xr = x + (size_t)n * kC;
  for (int c = lane; c < kC; c += 32) {
    float xv = xr[c];
    #pragma unroll
    for (int e = 0; e < kEtot; ++e) p[e] += xv * s_rw[e * kC + c];
  }
  #pragma unroll
  for (int e = 0; e < kEtot; ++e) {
    float v = p[e];
    for (int off = 16; off > 0; off >>= 1) v += __shfl_down(v, off);
    p[e] = v;
  }
  if (lane == 0) {
    float m = p[0];
    #pragma unroll
    for (int e = 1; e < kEtot; ++e) m = fmaxf(m, p[e]);
    float s = 0.f;
    float w[kEtot];
    #pragma unroll
    for (int e = 0; e < kEtot; ++e) { w[e] = __expf(p[e] - m); s += w[e]; }
    float inv = 1.f / s;
    #pragma unroll
    for (int e = 0; e < kEtot; ++e) { w[e] *= inv; rw_out[(size_t)n * kEtot + e] = w[e]; }
    // top-2, lowest index wins ties (matches jax.lax.top_k)
    int i0 = 0;
    #pragma unroll
    for (int e = 1; e < kEtot; ++e) if (w[e] > w[i0]) i0 = e;
    int i1 = (i0 == 0) ? 1 : 0;
    #pragma unroll
    for (int e = 0; e < kEtot; ++e) if (e != i0 && w[e] > w[i1]) i1 = e;
    float w0 = w[i0], w1 = w[i1];
    float wn = 1.f / (w0 + w1 + 1e-10f);
    w0 *= wn; w1 *= wn;
    meta_idx[n] = i0 | (i1 << 8);
    meta_w[2 * n]     = w0;
    meta_w[2 * n + 1] = w1;
    if (i0 < kEmlp) {
      int pos = atomicAdd(&counts[i0], 1);
      list_packed[i0 * kN + pos] = (n << 1);
      list_w[i0 * kN + pos] = w0;
    }
    if (i1 < kEmlp) {
      int pos = atomicAdd(&counts[i1], 1);
      list_packed[i1 * kN + pos] = (n << 1) | 1;
      list_w[i1 * kN + pos] = w1;
    }
  }
}

// ---------------------------------------------------------------------------
// 2) f32 -> bf16 conversion of x / fc_w / proj_w (bandwidth-bound, RNE)
// ---------------------------------------------------------------------------
__global__ __launch_bounds__(256) void convert_bf16_kernel(
    const float* __restrict__ x, const float* __restrict__ fc,
    const float* __restrict__ proj,
    unsigned short* __restrict__ xb, unsigned short* __restrict__ fcb,
    unsigned short* __restrict__ projb)
{
  const size_t stride = (size_t)gridDim.x * blockDim.x;
  const size_t t0 = (size_t)blockIdx.x * blockDim.x + threadIdx.x;
  const size_t nx = (size_t)kN * kC / 4;
  const size_t nw = (size_t)kEmlp * kH * kC / 4;
  for (size_t i = t0; i < nx; i += stride) {
    float4 v = ((const float4*)x)[i];
    ushort4 o; o.x = f32_to_bf16_rne(v.x); o.y = f32_to_bf16_rne(v.y);
    o.z = f32_to_bf16_rne(v.z); o.w = f32_to_bf16_rne(v.w);
    ((ushort4*)xb)[i] = o;
  }
  for (size_t i = t0; i < nw; i += stride) {
    float4 v = ((const float4*)fc)[i];
    ushort4 o; o.x = f32_to_bf16_rne(v.x); o.y = f32_to_bf16_rne(v.y);
    o.z = f32_to_bf16_rne(v.z); o.w = f32_to_bf16_rne(v.w);
    ((ushort4*)fcb)[i] = o;
  }
  for (size_t i = t0; i < nw; i += stride) {
    float4 v = ((const float4*)proj)[i];
    ushort4 o; o.x = f32_to_bf16_rne(v.x); o.y = f32_to_bf16_rne(v.y);
    o.z = f32_to_bf16_rne(v.z); o.w = f32_to_bf16_rne(v.w);
    ((ushort4*)projb)[i] = o;
  }
}

// ---------------------------------------------------------------------------
// 3) Fused per-expert dual GEMM, 32-token tiles, 16 waves/block.
//    N-blocked 4x: one A fragment feeds 4 WMMAs (2 ds_load + 8 global_load
//    per 4 v_wmma_f32_16x16x32_bf16). hidden = relu(x@fcT)^2 lives in LDS
//    as bf16; out = hidden @ projT, routing-weighted, scattered to per-
//    (token,slot) buffers (deterministic, no float atomics).
// ---------------------------------------------------------------------------
__global__ __launch_bounds__(512) void moe_mlp_kernel(
    const unsigned short* __restrict__ xb,
    const unsigned short* __restrict__ fcb,
    const unsigned short* __restrict__ projb,
    const int* __restrict__ counts,
    const int* __restrict__ list_packed,
    const float* __restrict__ list_w,
    float* __restrict__ slot_out,
    float* __restrict__ dummy)
{
  extern __shared__ char smem[];
  unsigned short* xs = (unsigned short*)smem;          // kMt * kXS halfs
  unsigned short* hs = xs + kMt * kXS;                 // kMt * kHS halfs
  float* swt = (float*)(hs + kMt * kHS);               // kMt
  unsigned long long* basep = (unsigned long long*)(swt + kMt); // kMt
  int* toks = (int*)(basep + kMt);                     // kMt

  const int e = blockIdx.y;
  const int cnt = counts[e];
  const int tbase = blockIdx.x * kMt;
  if (tbase >= cnt) return;
  const int tid = threadIdx.x;

  if (tid < kMt) {
    int li = tbase + tid;
    if (li < cnt) {
      int pk = list_packed[e * kN + li];
      int tok = pk >> 1, slot = pk & 1;
      toks[tid] = tok;
      swt[tid] = list_w[e * kN + li];
      basep[tid] = (unsigned long long)(slot_out + ((size_t)tok * 2 + slot) * kC);
    } else {
      toks[tid] = 0;
      swt[tid] = 0.f;
      basep[tid] = (unsigned long long)dummy;   // scratch sink, never read
    }
  }
  __syncthreads();

  // Gather 32 token rows of x (bf16) into padded LDS
  for (int i = tid; i < kMt * (kC / 8); i += 512) {
    int r = i / (kC / 8);
    int ch = i % (kC / 8);
    uint4 v = *(const uint4*)(xb + (size_t)toks[r] * kC + (size_t)ch * 8);
    *(uint4*)(xs + r * kXS + ch * 8) = v;
  }
  __syncthreads();

  const int wv   = tid >> 5;      // 16 waves
  const int lane = tid & 31;
  const int g    = lane >> 4;     // half-wave group (ISA A/B K-split)
  const int hl   = lane & 15;
  const int mt   = wv & 1;        // row sub-tile (16 rows each)
  const int wcol = wv >> 1;       // 8 column-wave groups

  // ---- GEMM1: hidden[32, H] = relu(x @ fcT)^2 ----
  const unsigned short* arow1 = xs + (mt * 16 + hl) * kXS + g * 8;
  #pragma unroll
  for (int jj = 0; jj < 4; ++jj) {
    const int ntb = wcol * 16 + jj * 4;               // 4 consecutive N-tiles
    v8f acc[4] = {};
    const unsigned short* wr0 =
        fcb + ((size_t)e * kH + (size_t)ntb * 16 + hl) * kC + g * 8;
    #pragma unroll 2
    for (int kt = 0; kt < kC / 32; ++kt) {
      const int k0 = kt * 32;
      FragBF a;
      a.q[0] = *(const uint4*)(arow1 + k0);           // K = k0+8g  .. +7
      a.q[1] = *(const uint4*)(arow1 + k0 + 16);      // K = k0+16+8g .. +7
      #pragma unroll
      for (int t = 0; t < 4; ++t) {
        FragBF b;
        b.q[0] = *(const uint4*)(wr0 + (size_t)t * 16 * kC + k0);
        b.q[1] = *(const uint4*)(wr0 + (size_t)t * 16 * kC + k0 + 16);
        acc[t] = __builtin_amdgcn_wmma_f32_16x16x32_bf16(
            false, a.v, false, b.v, (short)0, acc[t], false, false);
      }
    }
    #pragma unroll
    for (int t = 0; t < 4; ++t) {
      #pragma unroll
      for (int r = 0; r < 8; ++r) {
        int m = mt * 16 + r + 8 * g;                  // C/D: M = r + 8*(lane/16)
        float v = fmaxf(acc[t][r], 0.f);
        v = v * v;                                    // relu^2
        union { float f; unsigned u; } cv; cv.f = v;
        hs[m * kHS + (ntb + t) * 16 + hl] =
            (unsigned short)((cv.u + 0x8000u) >> 16); // cheap round-half-up
      }
    }
  }
  __syncthreads();

  // ---- GEMM2: out[32, C] = hidden @ projT ----
  const unsigned short* arow2 = hs + (mt * 16 + hl) * kHS + g * 8;
  #pragma unroll
  for (int jj = 0; jj < 2; ++jj) {
    const int ntb = wcol * 8 + jj * 4;
    v8f acc[4] = {};
    const unsigned short* wr0 =
        projb + ((size_t)e * kC + (size_t)ntb * 16 + hl) * kH + g * 8;
    #pragma unroll 2
    for (int kt = 0; kt < kH / 32; ++kt) {
      const int k0 = kt * 32;
      FragBF a;
      a.q[0] = *(const uint4*)(arow2 + k0);
      a.q[1] = *(const uint4*)(arow2 + k0 + 16);
      #pragma unroll
      for (int t = 0; t < 4; ++t) {
        FragBF b;
        b.q[0] = *(const uint4*)(wr0 + (size_t)t * 16 * kH + k0);
        b.q[1] = *(const uint4*)(wr0 + (size_t)t * 16 * kH + k0 + 16);
        acc[t] = __builtin_amdgcn_wmma_f32_16x16x32_bf16(
            false, a.v, false, b.v, (short)0, acc[t], false, false);
      }
    }
    #pragma unroll
    for (int t = 0; t < 4; ++t) {
      #pragma unroll
      for (int r = 0; r < 8; ++r) {
        int m = mt * 16 + r + 8 * g;
        float* op = (float*)basep[m];
        op[(ntb + t) * 16 + hl] = acc[t][r] * swt[m]; // pre-weighted slot write
      }
    }
  }
}

// ---------------------------------------------------------------------------
// 4) Combine: MLP slot buffers + weighted vocab-embedding gather -> output
// ---------------------------------------------------------------------------
__global__ __launch_bounds__(256) void combine_kernel(
    const float* __restrict__ ve_tables, const int* __restrict__ token_ids,
    const int* __restrict__ meta_idx, const float* __restrict__ meta_w,
    const float* __restrict__ slot_out, float* __restrict__ out)
{
  const int n = blockIdx.x;
  const int c4 = threadIdx.x;                      // kC/4 == 256 float4 per token
  const int mi = meta_idx[n];
  const int i0 = mi & 0xff;
  const int i1 = (mi >> 8) & 0xff;
  const float w0 = meta_w[2 * n];
  const float w1 = meta_w[2 * n + 1];
  const long tok = (long)token_ids[n];

  float4 acc = make_float4(0.f, 0.f, 0.f, 0.f);
  if (i0 < kEmlp) {
    float4 v = ((const float4*)(slot_out + (size_t)n * 2 * kC))[c4];
    acc.x += v.x; acc.y += v.y; acc.z += v.z; acc.w += v.w;
  } else {
    float4 v = ((const float4*)(ve_tables + (((size_t)(i0 - kEmlp)) * kV + tok) * kC))[c4];
    acc.x += w0 * v.x; acc.y += w0 * v.y; acc.z += w0 * v.z; acc.w += w0 * v.w;
  }
  if (i1 < kEmlp) {
    float4 v = ((const float4*)(slot_out + ((size_t)n * 2 + 1) * kC))[c4];
    acc.x += v.x; acc.y += v.y; acc.z += v.z; acc.w += v.w;
  } else {
    float4 v = ((const float4*)(ve_tables + (((size_t)(i1 - kEmlp)) * kV + tok) * kC))[c4];
    acc.x += w1 * v.x; acc.y += w1 * v.y; acc.z += w1 * v.z; acc.w += w1 * v.w;
  }
  ((float4*)(out + (size_t)n * kC))[c4] = acc;
}

// ---------------------------------------------------------------------------
extern "C" void kernel_launch(void* const* d_in, const int* in_sizes, int n_in,
                              void* d_out, int out_size, void* d_ws, size_t ws_size,
                              hipStream_t stream) {
  const float* x         = (const float*)d_in[0];
  const int*   token_ids = (const int*)d_in[1];
  const float* router_w  = (const float*)d_in[2];
  const float* fc_w      = (const float*)d_in[3];
  const float* proj_w    = (const float*)d_in[4];
  const float* ve_tables = (const float*)d_in[5];
  float* out    = (float*)d_out;
  float* rw_out = out + (size_t)kN * kC;   // second tuple output: routing_weights

  char* ws = (char*)d_ws;
  auto align256 = [](size_t v) { return (v + 255) & ~(size_t)255; };
  size_t o = 0;
  unsigned short* xb    = (unsigned short*)(ws + o); o = align256(o + (size_t)kN * kC * 2);
  unsigned short* fcb   = (unsigned short*)(ws + o); o = align256(o + (size_t)kEmlp * kH * kC * 2);
  unsigned short* projb = (unsigned short*)(ws + o); o = align256(o + (size_t)kEmlp * kC * kH * 2);
  float* slot_out       = (float*)(ws + o);          o = align256(o + (size_t)kN * 2 * kC * 4);
  float* dummy          = (float*)(ws + o);          o = align256(o + (size_t)kC * 4);
  int*   counts         = (int*)(ws + o);            o = align256(o + (size_t)kEmlp * 4);
  int*   list_packed    = (int*)(ws + o);            o = align256(o + (size_t)kEmlp * kN * 4);
  float* list_w         = (float*)(ws + o);          o = align256(o + (size_t)kEmlp * kN * 4);
  int*   meta_idx       = (int*)(ws + o);            o = align256(o + (size_t)kN * 4);
  float* meta_w         = (float*)(ws + o);          o = align256(o + (size_t)kN * 2 * 4);

  hipMemsetAsync(counts, 0, kEmlp * sizeof(int), stream);

  router_kernel<<<kN / 8, 256, 0, stream>>>(x, router_w, rw_out, counts,
                                            list_packed, list_w, meta_idx, meta_w);
  convert_bf16_kernel<<<2048, 256, 0, stream>>>(x, fc_w, proj_w, xb, fcb, projb);

  const size_t lds = (size_t)kMt * kXS * 2 + (size_t)kMt * kHS * 2
                   + kMt * sizeof(float) + kMt * sizeof(unsigned long long)
                   + kMt * sizeof(int);
  moe_mlp_kernel<<<dim3(kN / kMt, kEmlp), 512, lds, stream>>>(
      xb, fcb, projb, counts, list_packed, list_w, slot_out, dummy);

  combine_kernel<<<kN, 256, 0, stream>>>(ve_tables, token_ids, meta_idx, meta_w,
                                         slot_out, out);
  (void)in_sizes; (void)n_in; (void)out_size; (void)ws_size;
}
